// MoEMLPBase_79688823210283
// MI455X (gfx1250) — compile-verified, matrix operand order checked by reading, not run on hardware
//
#include <hip/hip_runtime.h>

// ---------------- problem constants ----------------
constexpr int BB = 4, SS = 2048, DD = 1024, FF = 4096, EE = 8;
constexpr int TT = BB * SS;          // 8192 tokens

// ---------------- MoE FFN tile config ----------------
constexpr int MT    = 32;            // tokens per block (2 WMMA M-tiles)
constexpr int NFC   = 512;           // F chunk held in LDS as h
constexpr int KSTEP = 32;            // WMMA K per step (16x16x32 bf16)

constexpr int XS_STRIDE = DD + 8;    // 1032
constexpr int HS_STRIDE = NFC + 8;   // 520
constexpr int WB_STRIDE = KSTEP + 8; // 40 (fallback: transposed weight tile [N][K])

constexpr int XS_BYTES  = MT * XS_STRIDE * 2;   // 66048
constexpr int HS_BYTES  = MT * HS_STRIDE * 2;   // 33280
constexpr int WB_BYTES  = NFC * WB_STRIDE * 2;  // 40960
constexpr int TOK_BYTES = 256;
constexpr int SMEM_V1 = TOK_BYTES + XS_BYTES + HS_BYTES + WB_BYTES; // 140544
constexpr int SMEM_V2 = TOK_BYTES + XS_BYTES + HS_BYTES;            // 99584

// ---------------- workspace layout ----------------
constexpr size_t W1T_OFF   = (size_t)1 << 20;                 // routing data below 1 MB
constexpr size_t W1T_BYTES = (size_t)EE * DD * FF * 2;        // 64 MB
constexpr size_t W2T_OFF   = W1T_OFF + W1T_BYTES;
constexpr size_t W2T_BYTES = W1T_BYTES;
constexpr size_t XB_OFF    = W2T_OFF + W2T_BYTES;
constexpr size_t XB_BYTES  = (size_t)TT * DD * 2;             // 16 MB
constexpr size_t WS_NEEDED = XB_OFF + XB_BYTES;               // ~145 MB

// ---------------- types ----------------
typedef __bf16 bf16;
typedef bf16  v2bf  __attribute__((ext_vector_type(2)));
typedef bf16  v4bf  __attribute__((ext_vector_type(4)));
typedef bf16  v8bf  __attribute__((ext_vector_type(8)));
typedef bf16  v16bf __attribute__((ext_vector_type(16)));
typedef float v8f   __attribute__((ext_vector_type(8)));

// fp32 -> bf16 round-to-nearest-even
__device__ __forceinline__ bf16 f2bf(float f) {
    unsigned u = __builtin_bit_cast(unsigned, f);
    u += 0x7fffu + ((u >> 16) & 1u);
    unsigned short h = (unsigned short)(u >> 16);
    return __builtin_bit_cast(bf16, h);
}

__device__ __forceinline__ v2bf f2bf2(float a, float b) {
    v2bf r; r[0] = f2bf(a); r[1] = f2bf(b); return r;
}

// branch-free tanh-GELU: v_exp_f32 + v_rcp_f32, no EXEC divergence
__device__ __forceinline__ float gelu_tanh(float v) {
    float z  = 0.7978845608028654f * v * (1.0f + 0.044715f * v * v);
    float az = __builtin_fabsf(z);
    float t  = __builtin_amdgcn_exp2f(az * -2.8853900817779268f); // exp(-2|z|)
    float th = (1.0f - t) * __builtin_amdgcn_rcpf(1.0f + t);      // tanh(|z|)
    th = __builtin_copysignf(th, z);
    return 0.5f * v * (1.0f + th);
}

// A/B fragment from a row-major bf16 tile with K contiguous (LDS or global):
// lanes 0-15: row lane,    K {kb..kb+7, kb+16..kb+23}
// lanes16-31: row lane-16, K {kb+8..kb+15, kb+24..kb+31}
template <typename PT>
__device__ __forceinline__ v16bf load_frag_t(const PT* base, size_t stride, int row,
                                             int kbase, int lane) {
    int r  = row + (lane & 15);
    int kb = kbase + ((lane >> 4) << 3);
    const v8bf* p0 = (const v8bf*)(base + (size_t)r * stride + kb);
    const v8bf* p1 = (const v8bf*)(base + (size_t)r * stride + kb + 16);
    v8bf a = *p0, b = *p1;
    return __builtin_shufflevector(a, b, 0, 1, 2, 3, 4, 5, 6, 7,
                                         8, 9, 10, 11, 12, 13, 14, 15);
}
__device__ __forceinline__ v16bf load_frag(const bf16* base, int stride, int row,
                                           int kbase, int lane) {
    return load_frag_t(base, (size_t)stride, row, kbase, lane);
}

#define WMMA_BF16(A, B, C) \
    __builtin_amdgcn_wmma_f32_16x16x32_bf16(false, (A), false, (B), (short)0, (C), false, false)

// ---------------- kernel 0: zero output + expert counters ----------------
__global__ void init_kernel(float* __restrict__ out, int* __restrict__ counts) {
    size_t n = (size_t)TT * DD;
    size_t idx = (size_t)blockIdx.x * blockDim.x + threadIdx.x;
    for (size_t i = idx; i < n; i += (size_t)gridDim.x * blockDim.x) out[i] = 0.0f;
    if (idx < EE) counts[idx] = 0;
}

// ---------------- kernel 1: router (softmax + top-2 + bucket append) ---------
__global__ void router_kernel(const float* __restrict__ x, const float* __restrict__ rw,
                              int* __restrict__ counts, int* __restrict__ btok,
                              float* __restrict__ bwt) {
    __shared__ float rws[EE * DD];   // 32 KB
    int tid = threadIdx.x, lane = tid & 31, wid = tid >> 5;
    for (int i = tid; i < EE * DD; i += 256) rws[i] = rw[i];
    __syncthreads();

    int t = blockIdx.x * 8 + wid;    // one wave32 per token
    float acc[EE];
#pragma unroll
    for (int e = 0; e < EE; ++e) acc[e] = 0.0f;
    for (int i = 0; i < DD / 32; ++i) {
        float xv = x[(size_t)t * DD + i * 32 + lane];
#pragma unroll
        for (int e = 0; e < EE; ++e) acc[e] += xv * rws[e * DD + i * 32 + lane];
    }
#pragma unroll
    for (int e = 0; e < EE; ++e)
        for (int off = 16; off; off >>= 1) acc[e] += __shfl_xor(acc[e], off, 32);

    if (lane == 0) {
        float mx = acc[0];
#pragma unroll
        for (int e = 1; e < EE; ++e) mx = fmaxf(mx, acc[e]);
        float p[EE], s = 0.0f;
#pragma unroll
        for (int e = 0; e < EE; ++e) { p[e] = __expf(acc[e] - mx); s += p[e]; }
        float inv = 1.0f / s;
#pragma unroll
        for (int e = 0; e < EE; ++e) p[e] *= inv;
        int i0 = 0; float p0 = p[0];
#pragma unroll
        for (int e = 1; e < EE; ++e) if (p[e] > p0) { p0 = p[e]; i0 = e; }
        int i1 = (i0 == 0) ? 1 : 0; float p1 = p[i1];
#pragma unroll
        for (int e = 0; e < EE; ++e)
            if (e != i0 && p[e] > p1) { p1 = p[e]; i1 = e; }
        float rn = 1.0f / (p0 + p1 + 1e-8f);
        int pos0 = atomicAdd(&counts[i0], 1);
        btok[i0 * TT + pos0] = t; bwt[i0 * TT + pos0] = p0 * rn;
        int pos1 = atomicAdd(&counts[i1], 1);
        btok[i1 * TT + pos1] = t; bwt[i1 * TT + pos1] = p1 * rn;
    }
}

// ---------------- kernel 1b: fp32 [R][C] -> transposed bf16 [C][R] -----------
__global__ void transpose_convert(const float* __restrict__ in, bf16* __restrict__ out,
                                  int R, int C) {
    __shared__ float tile[32][33];
    size_t ebase = (size_t)blockIdx.z * R * C;
    int r0 = blockIdx.y * 32, c0 = blockIdx.x * 32;
    int t = threadIdx.x;
    int rl = t >> 3, cl = (t & 7) * 4;            // read 32 rows x 8 float4 (coalesced)
    const float4 v = *(const float4*)(in + ebase + (size_t)(r0 + rl) * C + c0 + cl);
    tile[rl][cl + 0] = v.x; tile[rl][cl + 1] = v.y;
    tile[rl][cl + 2] = v.z; tile[rl][cl + 3] = v.w;
    __syncthreads();
    int cw = t >> 3, rw = (t & 7) * 4;            // write 32 out-rows x 8 b64 (coalesced)
    v4bf o = { f2bf(tile[rw + 0][cw]), f2bf(tile[rw + 1][cw]),
               f2bf(tile[rw + 2][cw]), f2bf(tile[rw + 3][cw]) };
    *(v4bf*)(out + ebase + (size_t)(c0 + cw) * R + r0 + rw) = o;
}

// ---------------- kernel 1c: x fp32 -> bf16 ----------------
__global__ void convert_x(const float* __restrict__ x, bf16* __restrict__ xb) {
    size_t n4 = (size_t)TT * DD / 4;
    for (size_t i = (size_t)blockIdx.x * blockDim.x + threadIdx.x; i < n4;
         i += (size_t)gridDim.x * blockDim.x) {
        const float4 v = ((const float4*)x)[i];
        v4bf o = { f2bf(v.x), f2bf(v.y), f2bf(v.z), f2bf(v.w) };
        ((v4bf*)xb)[i] = o;
    }
}

// ============ kernel 2a (big workspace): pre-converted transposed weights ====
// W1t: [E][F][D] bf16 (K=D contiguous), W2t: [E][D][F] bf16 (K=F contiguous).
// B fragments load straight from global (L2-resident, 16 MB/expert).
// No weight staging, no barriers inside the k-loops.
__launch_bounds__(256, 1)
__global__ void moe_ffn_v2(const bf16* __restrict__ xb,
                           const bf16* __restrict__ w1t, const float* __restrict__ b1,
                           const bf16* __restrict__ w2t, const float* __restrict__ b2,
                           const int* __restrict__ counts,
                           const int* __restrict__ btok, const float* __restrict__ bw,
                           float* __restrict__ out) {
    extern __shared__ __align__(16) char smem[];
    int*   tok_s = (int*)smem;
    float* twt_s = (float*)(smem + 128);
    bf16*  xs    = (bf16*)(smem + TOK_BYTES);                // [32][1032]
    bf16*  hs    = (bf16*)(smem + TOK_BYTES + XS_BYTES);     // [32][520]

    const int chunks = TT / MT;
    int e = blockIdx.x / chunks;
    int c = blockIdx.x % chunks;
    int cnt = counts[e];
    if (c * MT >= cnt) return;

    int tid = threadIdx.x, lane = tid & 31, wid = tid >> 5;

    if (tid < MT) {
        int g = c * MT + tid;
        if (g < cnt) { tok_s[tid] = btok[e * TT + g]; twt_s[tid] = bw[e * TT + g]; }
        else         { tok_s[tid] = 0;               twt_s[tid] = 0.0f; }
    }
    __syncthreads();

    // gather 32 token rows (bf16 -> pure b128 copy, no conversion)
#pragma unroll 4
    for (int i = tid; i < MT * (DD / 8); i += 256) {
        int r = i >> 7, c8 = i & 127;
        v8bf v = *(const v8bf*)(xb + (size_t)tok_s[r] * DD + c8 * 8);
        *(v8bf*)(xs + r * XS_STRIDE + c8 * 8) = v;
    }
    __syncthreads();

    const bf16* w1e = w1t + (size_t)e * FF * DD;
    const bf16* w2e = w2t + (size_t)e * DD * FF;

    v8f acc2[2][8];                              // y: [Mtile][Ntile], 128 cols/wave
#pragma unroll
    for (int m = 0; m < 2; ++m)
#pragma unroll
        for (int n = 0; n < 8; ++n) acc2[m][n] = {};

    for (int fc = 0; fc < FF / NFC; ++fc) {
        int f0 = fc * NFC;

        // ---- phase 1: h[32][512] = xs @ W1[:, f0:f0+512] + b1 (no barriers) ----
        v8f acc1[2][4];
#pragma unroll
        for (int nt = 0; nt < 4; ++nt) {
            float bv = b1[(size_t)e * FF + f0 + wid * 64 + nt * 16 + (lane & 15)];
            v8f ci;
#pragma unroll
            for (int q = 0; q < 8; ++q) ci[q] = bv;
            acc1[0][nt] = ci; acc1[1][nt] = ci;
        }
        for (int ks = 0; ks < DD / KSTEP; ++ks) {
            v16bf a0 = load_frag(xs, XS_STRIDE, 0,  ks * KSTEP, lane);
            v16bf a1 = load_frag(xs, XS_STRIDE, 16, ks * KSTEP, lane);
#pragma unroll
            for (int nt = 0; nt < 4; ++nt) {
                v16bf bf = load_frag_t(w1e, (size_t)DD, f0 + wid * 64 + nt * 16,
                                       ks * KSTEP, lane);   // 2x global_load_b128
                acc1[0][nt] = WMMA_BF16(a0, bf, acc1[0][nt]);
                acc1[1][nt] = WMMA_BF16(a1, bf, acc1[1][nt]);
            }
        }
        // GELU -> hs
#pragma unroll
        for (int mt = 0; mt < 2; ++mt)
#pragma unroll
            for (int nt = 0; nt < 4; ++nt) {
                int n = wid * 64 + nt * 16 + (lane & 15);
#pragma unroll
                for (int q = 0; q < 8; q += 2) {
                    int m = mt * 16 + q + ((lane >> 4) << 3);
                    v2bf p = f2bf2(gelu_tanh(acc1[mt][nt][q]),
                                   gelu_tanh(acc1[mt][nt][q + 1]));
                    hs[m * HS_STRIDE + n]       = p[0];
                    hs[(m + 1) * HS_STRIDE + n] = p[1];
                }
            }
        __syncthreads();

        // ---- phase 2: y += hs @ W2[f0:f0+512, :] (no barriers in k-loop) ----
        for (int ks = 0; ks < NFC / KSTEP; ++ks) {
            v16bf a0 = load_frag(hs, HS_STRIDE, 0,  ks * KSTEP, lane);
            v16bf a1 = load_frag(hs, HS_STRIDE, 16, ks * KSTEP, lane);
#pragma unroll
            for (int nt = 0; nt < 8; ++nt) {
                v16bf bf = load_frag_t(w2e, (size_t)FF, wid * 128 + nt * 16,
                                       f0 + ks * KSTEP, lane);
                acc2[0][nt] = WMMA_BF16(a0, bf, acc2[0][nt]);
                acc2[1][nt] = WMMA_BF16(a1, bf, acc2[1][nt]);
            }
        }
        __syncthreads();   // protect hs before next f-chunk's GELU overwrite
    }

    // epilogue: out[token] += gate_w * (y + b2)
#pragma unroll
    for (int mt = 0; mt < 2; ++mt)
#pragma unroll
        for (int nt = 0; nt < 8; ++nt) {
            int col = wid * 128 + nt * 16 + (lane & 15);
            float b2v = b2[(size_t)e * DD + col];
#pragma unroll
            for (int q = 0; q < 8; ++q) {
                int m = mt * 16 + q + ((lane >> 4) << 3);
                float val = (acc2[mt][nt][q] + b2v) * twt_s[m];
                atomicAdd(out + (size_t)tok_s[m] * DD + col, val);
            }
        }
}

// ============ kernel 2b (small workspace fallback): fused convert+stage ======
__launch_bounds__(256, 1)
__global__ void moe_ffn_v1(const float* __restrict__ x,
                           const float* __restrict__ W1, const float* __restrict__ b1,
                           const float* __restrict__ W2, const float* __restrict__ b2,
                           const int* __restrict__ counts,
                           const int* __restrict__ btok, const float* __restrict__ bw,
                           float* __restrict__ out) {
    extern __shared__ __align__(16) char smem[];
    int*   tok_s = (int*)smem;
    float* twt_s = (float*)(smem + 128);
    bf16*  xs    = (bf16*)(smem + TOK_BYTES);
    bf16*  hs    = (bf16*)(smem + TOK_BYTES + XS_BYTES);
    bf16*  wb    = (bf16*)(smem + TOK_BYTES + XS_BYTES + HS_BYTES);

    const int chunks = TT / MT;
    int e = blockIdx.x / chunks;
    int c = blockIdx.x % chunks;
    int cnt = counts[e];
    if (c * MT >= cnt) return;

    int tid = threadIdx.x, lane = tid & 31, wid = tid >> 5;

    if (tid < MT) {
        int g = c * MT + tid;
        if (g < cnt) { tok_s[tid] = btok[e * TT + g]; twt_s[tid] = bw[e * TT + g]; }
        else         { tok_s[tid] = 0;               twt_s[tid] = 0.0f; }
    }
    __syncthreads();

#pragma unroll 4
    for (int i = tid; i < MT * (DD / 4); i += 256) {
        int r = i >> 8, c4 = i & 255;
        const float4 v = *(const float4*)(x + (size_t)tok_s[r] * DD + c4 * 4);
        v2bf p0 = f2bf2(v.x, v.y), p1 = f2bf2(v.z, v.w);
        v4bf pv = {p0[0], p0[1], p1[0], p1[1]};
        *(v4bf*)(xs + r * XS_STRIDE + c4 * 4) = pv;
    }
    __syncthreads();

    v8f acc2[2][2][4];
#pragma unroll
    for (int h = 0; h < 2; ++h)
#pragma unroll
        for (int m = 0; m < 2; ++m)
#pragma unroll
            for (int n = 0; n < 4; ++n) acc2[h][m][n] = {};

    for (int fc = 0; fc < FF / NFC; ++fc) {
        int f0 = fc * NFC;
        v8f acc1[2][4];
#pragma unroll
        for (int nt = 0; nt < 4; ++nt) {
            float bv = b1[(size_t)e * FF + f0 + wid * 64 + nt * 16 + (lane & 15)];
            v8f ci;
#pragma unroll
            for (int q = 0; q < 8; ++q) ci[q] = bv;
            acc1[0][nt] = ci; acc1[1][nt] = ci;
        }
        for (int ks = 0; ks < DD / KSTEP; ++ks) {
#pragma unroll 4
            for (int i = 0; i < 16; ++i) {
                int L = tid + 256 * i;
                int dl = L >> 7, fq = L & 127;
                const float* src = W1 + (size_t)e * DD * FF +
                                   (size_t)(ks * KSTEP + dl) * FF + f0 + fq * 4;
                const float4 v = *(const float4*)src;
                if (((fq & 7) == 0) && (ks + 1 < DD / KSTEP))
                    __builtin_prefetch(src + (size_t)KSTEP * FF, 0, 1);
                int n = fq * 4;
                v2bf p0 = f2bf2(v.x, v.y), p1 = f2bf2(v.z, v.w);
                wb[(n + 0) * WB_STRIDE + dl] = p0[0];
                wb[(n + 1) * WB_STRIDE + dl] = p0[1];
                wb[(n + 2) * WB_STRIDE + dl] = p1[0];
                wb[(n + 3) * WB_STRIDE + dl] = p1[1];
            }
            __syncthreads();
            v16bf a0 = load_frag(xs, XS_STRIDE, 0,  ks * KSTEP, lane);
            v16bf a1 = load_frag(xs, XS_STRIDE, 16, ks * KSTEP, lane);
#pragma unroll
            for (int nt = 0; nt < 4; ++nt) {
                v16bf bf = load_frag(wb, WB_STRIDE, wid * 64 + nt * 16, 0, lane);
                acc1[0][nt] = WMMA_BF16(a0, bf, acc1[0][nt]);
                acc1[1][nt] = WMMA_BF16(a1, bf, acc1[1][nt]);
            }
            __syncthreads();
        }
#pragma unroll
        for (int mt = 0; mt < 2; ++mt)
#pragma unroll
            for (int nt = 0; nt < 4; ++nt) {
                int n = wid * 64 + nt * 16 + (lane & 15);
#pragma unroll
                for (int q = 0; q < 8; q += 2) {
                    int m = mt * 16 + q + ((lane >> 4) << 3);
                    v2bf p = f2bf2(gelu_tanh(acc1[mt][nt][q]),
                                   gelu_tanh(acc1[mt][nt][q + 1]));
                    hs[m * HS_STRIDE + n]       = p[0];
                    hs[(m + 1) * HS_STRIDE + n] = p[1];
                }
            }
        __syncthreads();

        for (int ks = 0; ks < NFC / KSTEP; ++ks) {
            v16bf a0 = load_frag(hs, HS_STRIDE, 0,  ks * KSTEP, lane);
            v16bf a1 = load_frag(hs, HS_STRIDE, 16, ks * KSTEP, lane);
#pragma unroll
            for (int hh = 0; hh < 2; ++hh) {
#pragma unroll 4
                for (int i = 0; i < 16; ++i) {
                    int L = tid + 256 * i;
                    int fl = L >> 7, dq = L & 127;
                    const float* src = W2 + (size_t)e * FF * DD +
                                       (size_t)(f0 + ks * KSTEP + fl) * DD +
                                       hh * 512 + dq * 4;
                    const float4 v = *(const float4*)src;
                    if (((dq & 7) == 0) && (ks + 1 < NFC / KSTEP))
                        __builtin_prefetch(src + (size_t)KSTEP * DD, 0, 1);
                    int n = dq * 4;
                    v2bf p0 = f2bf2(v.x, v.y), p1 = f2bf2(v.z, v.w);
                    wb[(n + 0) * WB_STRIDE + fl] = p0[0];
                    wb[(n + 1) * WB_STRIDE + fl] = p0[1];
                    wb[(n + 2) * WB_STRIDE + fl] = p1[0];
                    wb[(n + 3) * WB_STRIDE + fl] = p1[1];
                }
                __syncthreads();
#pragma unroll
                for (int nt = 0; nt < 4; ++nt) {
                    v16bf bf = load_frag(wb, WB_STRIDE, wid * 64 + nt * 16, 0, lane);
                    acc2[hh][0][nt] = WMMA_BF16(a0, bf, acc2[hh][0][nt]);
                    acc2[hh][1][nt] = WMMA_BF16(a1, bf, acc2[hh][1][nt]);
                }
                __syncthreads();
            }
        }
    }

#pragma unroll
    for (int hh = 0; hh < 2; ++hh)
#pragma unroll
        for (int mt = 0; mt < 2; ++mt)
#pragma unroll
            for (int nt = 0; nt < 4; ++nt) {
                int col = hh * 512 + wid * 64 + nt * 16 + (lane & 15);
                float b2v = b2[(size_t)e * DD + col];
#pragma unroll
                for (int q = 0; q < 8; ++q) {
                    int m = mt * 16 + q + ((lane >> 4) << 3);
                    float val = (acc2[hh][mt][nt][q] + b2v) * twt_s[m];
                    atomicAdd(out + (size_t)tok_s[m] * DD + col, val);
                }
            }
}

// ---------------- host-side launch ----------------
extern "C" void kernel_launch(void* const* d_in, const int* in_sizes, int n_in,
                              void* d_out, int out_size, void* d_ws, size_t ws_size,
                              hipStream_t stream) {
    const float* x        = (const float*)d_in[0];
    const float* router_w = (const float*)d_in[1];
    const float* W1       = (const float*)d_in[2];
    const float* b1       = (const float*)d_in[3];
    const float* W2       = (const float*)d_in[4];
    const float* b2       = (const float*)d_in[5];
    float* out = (float*)d_out;

    int*   counts = (int*)d_ws;
    int*   btok   = (int*)((char*)d_ws + 1024);
    float* bwt    = (float*)((char*)d_ws + 1024 + (size_t)EE * TT * sizeof(int));

    init_kernel<<<2048, 256, 0, stream>>>(out, counts);
    router_kernel<<<TT / 8, 256, 0, stream>>>(x, router_w, counts, btok, bwt);

    if (ws_size >= WS_NEEDED) {
        bf16* w1t = (bf16*)((char*)d_ws + W1T_OFF);   // [E][F][D]
        bf16* w2t = (bf16*)((char*)d_ws + W2T_OFF);   // [E][D][F]
        bf16* xb  = (bf16*)((char*)d_ws + XB_OFF);    // [T][D]
        convert_x<<<4096, 256, 0, stream>>>(x, xb);
        transpose_convert<<<dim3(FF / 32, DD / 32, EE), 256, 0, stream>>>(W1, w1t, DD, FF);
        transpose_convert<<<dim3(DD / 32, FF / 32, EE), 256, 0, stream>>>(W2, w2t, FF, DD);
        moe_ffn_v2<<<EE * (TT / MT), 256, SMEM_V2, stream>>>(xb, w1t, b1, w2t, b2,
                                                             counts, btok, bwt, out);
    } else {
        moe_ffn_v1<<<EE * (TT / MT), 256, SMEM_V1, stream>>>(x, W1, b1, W2, b2,
                                                             counts, btok, bwt, out);
    }
}
